// MultiScaleDeformableAttention_1_15307263443642
// MI455X (gfx1250) — matrix operand order repaired and checked by reading, not compiled, not run on hardware
//
#include <hip/hip_runtime.h>
#include <hip/hip_bf16.h>

// ---------------------------------------------------------------------------
// Multi-Scale Deformable Attention for MI455X (gfx1250, wave32, WMMA + TDM).
// H=8 L=4 P=2 D=32 C=256, B=4 NQ=1000, NV=21760, SHAPES square {128,64,32,16}.
// ---------------------------------------------------------------------------

typedef __attribute__((ext_vector_type(16))) _Float16 v16h;
typedef __attribute__((ext_vector_type(8)))  _Float16 v8h;
typedef __attribute__((ext_vector_type(8)))  float    v8f;

typedef __attribute__((ext_vector_type(4))) unsigned int tdm_v4u;
typedef __attribute__((ext_vector_type(8))) int          tdm_v8i;
typedef __attribute__((ext_vector_type(4))) int          tdm_v4i;

#define C_DIM   256
#define H_DIM   8
#define D_DIM   32
#define NV_TOT  21760
#define NQ_DIM  1000
#define B_DIM   4

#if defined(__has_builtin)
#if __has_builtin(__builtin_amdgcn_tensor_load_to_lds) && \
    __has_builtin(__builtin_amdgcn_s_wait_tensorcnt)
#define HAVE_TDM 1
#endif
#endif

// ---------------------------------------------------------------------------
// One-time prep: Wt[n][k] = (f16) W[k][n]   (transpose + downconvert).
// Lets the GEMM stage its LDS B-tile with a single Tensor-Data-Mover DMA.
// ---------------------------------------------------------------------------
__global__ __launch_bounds__(256) void weight_transpose_f16_kernel(
    const float* __restrict__ W, int Nw, _Float16* __restrict__ Wt)
{
    const int idx = blockIdx.x * 256 + threadIdx.x;   // over Nw*256
    const int n = idx >> 8;          // output row  (col of W)
    const int k = idx & 255;         // output col  (row of W)
    if (n < Nw) Wt[(size_t)n * C_DIM + k] = (_Float16)W[(size_t)k * Nw + n];
}

// ---------------------------------------------------------------------------
// Generic f32-in / f16-WMMA / f32-out GEMM:  Out = A[M x 256] * W[256 x Nw] + b
// Block tile 64(M) x 64(N), 256 threads = 8 waves, wave tile 16 x 32
// (two v_wmma_f32_16x16x32_f16 accumulators), K loop 256 in steps of 32.
// B-tile (pre-transposed f16 weights, 64 x 256 = 32 KB) is DMA'd into LDS by
// the Tensor Data Mover with pad_interval=128dw / pad_amount=4dw so the LDS
// image has a 528 B row pitch == WtLds[64][264] (conflict-free ds_load_b128).
// OUT_MODE 0: Out row-major [M x ldo].
// OUT_MODE 1: value-projection scatter to [B,H,NV,D].
// ---------------------------------------------------------------------------
template <int OUT_MODE>
__global__ __launch_bounds__(256) void gemm_wmma_f16(
    const float* __restrict__ A, int M,
    const _Float16* __restrict__ Wt,     // [Nw][256] f16, pre-transposed
    const float* __restrict__ bias,
    float* __restrict__ Out, int ldo)
{
    __shared__ _Float16 WtLds[64][C_DIM + 8];   // 64 x 264 halves = 33792 B

    const int m0 = blockIdx.x * 64;
    const int n0 = blockIdx.y * 64;

    const int wave = threadIdx.x >> 5;   // 0..7
    const int lane = threadIdx.x & 31;

#if HAVE_TDM
    // ---- TDM: one descriptor moves the whole 64x256 f16 slice ------------
    if (wave == 0) {
        const unsigned long long ga =
            (unsigned long long)(size_t)(Wt + (size_t)n0 * C_DIM);
        tdm_v4u g0;
        g0[0] = 1u;                                   // count=1 (valid D#)
        g0[1] = (unsigned int)(size_t)(&WtLds[0][0]); // LDS byte address
        g0[2] = (unsigned int)(ga & 0xFFFFFFFFull);   // global_addr[31:0]
        g0[3] = (unsigned int)((ga >> 32) & 0x1FFFFFFull) | 0x80000000u; // type=2
        tdm_v8i g1;
        g1[0] = 0x07910000;   // data_size=2B | pad_en | interval=128dw | amt=4dw
        g1[1] = 0x40000000;   // tensor_dim0 = 16384 (low16 in [31:16])
        g1[2] = 0x00010000;   // tensor_dim0 hi=0 | tensor_dim1 = 1
        g1[3] = 0x40000000;   // tensor_dim1 hi=0 | tile_dim0 = 16384
        g1[4] = 1;            // tile_dim1 = 1, tile_dim2 = 0
        g1[5] = 16384;        // tensor_dim0_stride (elements)
        g1[6] = 0;
        g1[7] = 0;
        tdm_v4i g2 = {0, 0, 0, 0};
        tdm_v4i g3 = {0, 0, 0, 0};
        tdm_v8i g4 = {0, 0, 0, 0, 0, 0, 0, 0};
        __builtin_amdgcn_tensor_load_to_lds(g0, g1, g2, g3, g4, 0);
        __builtin_amdgcn_s_wait_tensorcnt(0);         // drain TENSORcnt
    }
#else
    // ---- fallback: cooperative copy (already f16, already transposed) ----
    for (int idx = threadIdx.x; idx < 64 * C_DIM; idx += 256) {
        const int n = idx >> 8;
        const int k = idx & 255;
        WtLds[n][k] = Wt[((size_t)n0 + n) * C_DIM + k];
    }
#endif
    __syncthreads();

    const int wm   = wave >> 1;          // 0..3  -> M sub-tile
    const int wn   = wave & 1;           // 0..1  -> N sub-tile (32 wide)
    const int l15  = lane & 15;
    const bool hiHalf = (lane >= 16);

    // A fragment (16-bit A 16x32): lanes 0..15 hold K {0..7,16..23},
    // lanes 16..31 hold K {8..15,24..31}; row = m0 + 16*wm + (lane&15).
    int rowA = m0 + wm * 16 + l15;
    if (rowA > M - 1) rowA = M - 1;      // clamp for ragged M (stores guarded)
    const float* __restrict__ arow = A + (size_t)rowA * C_DIM + (hiHalf ? 8 : 0);

    // B fragment (16-bit B 32x16): lanes 0..15 hold K {0..7,16..23}, lanes 16..31
    // hold K {8..15,24..31}; lane%16 selects column.
    const int kbB = hiHalf ? 8 : 0;
    const _Float16* __restrict__ bcol0 = &WtLds[wn * 32 + l15][kbB];
    const _Float16* __restrict__ bcol1 = &WtLds[wn * 32 + 16 + l15][kbB];

    v8f acc0 = {};
    v8f acc1 = {};

#pragma unroll
    for (int k0 = 0; k0 < C_DIM; k0 += 32) {
        if (k0 + 32 < C_DIM)
            __builtin_prefetch(arow + k0 + 32, 0, 0);   // global_prefetch_b8

        // ---- A fragment: 4x float4 (b128 loads), convert to f16 ----------
        const float4* ap4 = (const float4*)(arow + k0);
        float4 a0 = ap4[0];              // K  0.. 3 (+kbase)
        float4 a1 = ap4[1];              // K  4.. 7
        float4 a2 = ap4[4];              // K 16..19
        float4 a3 = ap4[5];              // K 20..23
        v16h af;
        af[0]  = (_Float16)a0.x; af[1]  = (_Float16)a0.y;
        af[2]  = (_Float16)a0.z; af[3]  = (_Float16)a0.w;
        af[4]  = (_Float16)a1.x; af[5]  = (_Float16)a1.y;
        af[6]  = (_Float16)a1.z; af[7]  = (_Float16)a1.w;
        af[8]  = (_Float16)a2.x; af[9]  = (_Float16)a2.y;
        af[10] = (_Float16)a2.z; af[11] = (_Float16)a2.w;
        af[12] = (_Float16)a3.x; af[13] = (_Float16)a3.y;
        af[14] = (_Float16)a3.z; af[15] = (_Float16)a3.w;

        // ---- B fragments: two 8-half runs (K 0..7, 16..23, +kbB) from LDS ------------------
        v8h b0lo = *(const v8h*)(bcol0 + k0);
        v8h b0hi = *(const v8h*)(bcol0 + k0 + 16);
        v8h b1lo = *(const v8h*)(bcol1 + k0);
        v8h b1hi = *(const v8h*)(bcol1 + k0 + 16);
        v16h bf0 = __builtin_shufflevector(b0lo, b0hi,
            0,1,2,3,4,5,6,7,8,9,10,11,12,13,14,15);
        v16h bf1 = __builtin_shufflevector(b1lo, b1hi,
            0,1,2,3,4,5,6,7,8,9,10,11,12,13,14,15);

        acc0 = __builtin_amdgcn_wmma_f32_16x16x32_f16(
            false, af, false, bf0, (short)0, acc0, false, false);
        acc1 = __builtin_amdgcn_wmma_f32_16x16x32_f16(
            false, af, false, bf1, (short)0, acc1, false, false);
    }

    // ---- epilogue: bias + store ------------------------------------------
    // C/D layout: VGPR r, lane l: M = r + (l>=16 ? 8 : 0), N = l & 15.
    const int nloc0 = n0 + wn * 32 + l15;
    const int nloc1 = nloc0 + 16;
    const float bv0 = bias[nloc0];
    const float bv1 = bias[nloc1];
    const int mbase = m0 + wm * 16 + (hiHalf ? 8 : 0);

#pragma unroll
    for (int r = 0; r < 8; ++r) {
        const int mg = mbase + r;
        if (mg >= M) continue;
        const float o0 = acc0[r] + bv0;
        const float o1 = acc1[r] + bv1;
        if (OUT_MODE == 0) {
            Out[(size_t)mg * ldo + nloc0] = o0;
            Out[(size_t)mg * ldo + nloc1] = o1;
        } else {
            const int b  = mg / NV_TOT;
            const int nv = mg - b * NV_TOT;
            const int h0 = nloc0 >> 5, d0 = nloc0 & 31;
            const int h1 = nloc1 >> 5, d1 = nloc1 & 31;
            Out[(((size_t)b * H_DIM + h0) * NV_TOT + nv) * D_DIM + d0] = o0;
            Out[(((size_t)b * H_DIM + h1) * NV_TOT + nv) * D_DIM + d1] = o1;
        }
    }
}

// ---------------------------------------------------------------------------
// Fused softmax + bilinear sampling + attention-weighted reduce.
// One block per (b,q); wave = head h; lane = channel d (D == wave32 width).
// v: [B,H,NV,D], off: [M,128], logits: [M,64], outh: [M, H*D] = [M,256].
// ---------------------------------------------------------------------------
__global__ __launch_bounds__(256) void msda_sample_kernel(
    const float* __restrict__ v,
    const float* __restrict__ ref,      // [B,NQ,L,2]
    const float* __restrict__ off,
    const float* __restrict__ logits,
    float* __restrict__ outh)
{
    const int m    = blockIdx.x;        // b*NQ + q
    const int h    = threadIdx.x >> 5;
    const int lane = threadIdx.x & 31;
    const int b    = m / NQ_DIM;

    // softmax over L*P = 8 (addresses uniform in wave -> scalarizable loads)
    const float* lg = logits + (size_t)m * 64 + h * 8;
    float w[8];
    float mx = -1e30f;
#pragma unroll
    for (int i = 0; i < 8; ++i) { w[i] = lg[i]; mx = fmaxf(mx, w[i]); }
    float s = 0.f;
#pragma unroll
    for (int i = 0; i < 8; ++i) { w[i] = __expf(w[i] - mx); s += w[i]; }
    const float inv = 1.f / s;

    const int dimL[4]   = {128, 64, 32, 16};
    const int startL[4] = {0, 16384, 20480, 21504};

    const float* __restrict__ ofs = off + (size_t)m * 128 + h * 16;
    const float* __restrict__ rp  = ref + (size_t)m * 8;           // [L][2]
    const float* __restrict__ vbh = v + ((size_t)b * H_DIM + h) * NV_TOT * D_DIM;

    float acc = 0.f;
#pragma unroll
    for (int l = 0; l < 4; ++l) {
        const int   wl  = dimL[l];                 // square levels: Hl == Wl
        const float wlf = (float)wl;
        const float invwl = 1.f / wlf;
        const float rx = rp[l * 2 + 0];
        const float ry = rp[l * 2 + 1];
        const float* __restrict__ vl = vbh + (size_t)startL[l] * D_DIM;
#pragma unroll
        for (int p = 0; p < 2; ++p) {
            const float lx = rx + ofs[l * 4 + p * 2 + 0] * invwl;
            const float ly = ry + ofs[l * 4 + p * 2 + 1] * invwl;
            const float x = lx * wlf - 0.5f;
            const float y = ly * wlf - 0.5f;
            const float x0f = floorf(x);
            const float y0f = floorf(y);
            const float fx = x - x0f;
            const float fy = y - y0f;
            const int x0 = (int)x0f;
            const int y0 = (int)y0f;
            const float aw = w[l * 2 + p] * inv;

            const float cw[4] = { (1.f - fx) * (1.f - fy) * aw,
                                  fx         * (1.f - fy) * aw,
                                  (1.f - fx) * fy         * aw,
                                  fx         * fy         * aw };
            const int cx[4] = { x0, x0 + 1, x0,     x0 + 1 };
            const int cy[4] = { y0, y0,     y0 + 1, y0 + 1 };
#pragma unroll
            for (int c = 0; c < 4; ++c) {
                const int xi = cx[c], yi = cy[c];
                if (xi >= 0 && xi < wl && yi >= 0 && yi < wl) {
                    acc += cw[c] * vl[((size_t)(yi * wl + xi)) * D_DIM + lane];
                }
            }
        }
    }
    outh[(size_t)m * C_DIM + h * D_DIM + lane] = acc;
}

// ---------------------------------------------------------------------------
// Launch graph:
//   4x weight transpose/convert -> value-proj GEMM -> off GEMM -> logit GEMM
//   -> sampler -> out GEMM
// ---------------------------------------------------------------------------
extern "C" void kernel_launch(void* const* d_in, const int* in_sizes, int n_in,
                              void* d_out, int out_size, void* d_ws, size_t ws_size,
                              hipStream_t stream) {
    const float* query  = (const float*)d_in[0];
    const float* value  = (const float*)d_in[1];
    const float* refpts = (const float*)d_in[2];
    const float* W_off  = (const float*)d_in[3];
    const float* b_off  = (const float*)d_in[4];
    const float* W_attn = (const float*)d_in[5];
    const float* b_attn = (const float*)d_in[6];
    const float* W_val  = (const float*)d_in[7];
    const float* b_val  = (const float*)d_in[8];
    const float* W_out  = (const float*)d_in[9];
    const float* b_out  = (const float*)d_in[10];
    float* out = (float*)d_out;

    const int Mv = B_DIM * NV_TOT;      // 87040
    const int Mq = B_DIM * NQ_DIM;      // 4000

    float* v_ws    = (float*)d_ws;                      // [B,H,NV,D] = 87040*256
    float* off_ws  = v_ws   + (size_t)Mv * C_DIM;       // [Mq,128]
    float* log_ws  = off_ws + (size_t)Mq * 128;         // [Mq,64]
    float* outh_ws = log_ws + (size_t)Mq * 64;          // [Mq,256]
    _Float16* wt_val  = (_Float16*)(outh_ws + (size_t)Mq * C_DIM); // [256][256]
    _Float16* wt_off  = wt_val  + 256 * C_DIM;                     // [128][256]
    _Float16* wt_attn = wt_off  + 128 * C_DIM;                     // [ 64][256]
    _Float16* wt_out  = wt_attn +  64 * C_DIM;                     // [256][256]

    const dim3 blk(256);

    // 0) pre-transpose + f16-convert all weights (tiny, one-time per launch)
    weight_transpose_f16_kernel<<<dim3(256 * C_DIM / 256), blk, 0, stream>>>(
        W_val, 256, wt_val);
    weight_transpose_f16_kernel<<<dim3(128 * C_DIM / 256), blk, 0, stream>>>(
        W_off, 128, wt_off);
    weight_transpose_f16_kernel<<<dim3(64 * C_DIM / 256), blk, 0, stream>>>(
        W_attn, 64, wt_attn);
    weight_transpose_f16_kernel<<<dim3(256 * C_DIM / 256), blk, 0, stream>>>(
        W_out, 256, wt_out);

    // 1) v = (value @ W_val + b_val) scattered to [B,H,NV,D]
    gemm_wmma_f16<1><<<dim3(Mv / 64, C_DIM / 64), blk, 0, stream>>>(
        value, Mv, wt_val, b_val, v_ws, 0);

    // 2) off = query @ W_off + b_off            [Mq,128]
    gemm_wmma_f16<0><<<dim3((Mq + 63) / 64, 128 / 64), blk, 0, stream>>>(
        query, Mq, wt_off, b_off, off_ws, 128);

    // 3) attn logits = query @ W_attn + b_attn  [Mq,64]
    gemm_wmma_f16<0><<<dim3((Mq + 63) / 64, 1), blk, 0, stream>>>(
        query, Mq, wt_attn, b_attn, log_ws, 64);

    // 4) softmax + bilinear sample + reduce -> [Mq, H*D]
    msda_sample_kernel<<<dim3(Mq), blk, 0, stream>>>(
        v_ws, refpts, off_ws, log_ws, outh_ws);

    // 5) out = outh @ W_out + b_out             [Mq,256]
    gemm_wmma_f16<0><<<dim3((Mq + 63) / 64, C_DIM / 64), blk, 0, stream>>>(
        outh_ws, Mq, wt_out, b_out, out, 256);
}
